// Supernetwork_52742198395348
// MI455X (gfx1250) — matrix operand order repaired and checked by reading, not compiled, not run on hardware
//
#include <hip/hip_runtime.h>
#include <hip/hip_bf16.h>

typedef __attribute__((ext_vector_type(16))) _Float16 v16h;
typedef __attribute__((ext_vector_type(8)))  float    v8f;
typedef __attribute__((ext_vector_type(4)))  float    f32x4;

#define WMMA_F32_F16(a, b, c) \
  __builtin_amdgcn_wmma_f32_16x16x32_f16(false, (a), false, (b), (short)0, (c), false, false)

// ---------------------------------------------------------------------------
// Fold conv bias + batchnorm into per-channel scale/shift: y = conv*scale+shift
// ---------------------------------------------------------------------------
__global__ void bnfold_kernel(const float* __restrict__ g, const float* __restrict__ bb,
                              const float* __restrict__ m, const float* __restrict__ v,
                              const float* __restrict__ cb, float* __restrict__ ss) {
  int c = threadIdx.x;
  if (c >= 32) return;
  float inv = g[c] * rsqrtf(v[c] + 1e-5f);
  ss[c]      = inv;
  ss[32 + c] = cb[c] * inv + bb[c] - m[c] * inv;
}

// embWbh[o][j] = (emb[o] @ Wxh)[j] + bh[j]   (3 x 128)
__global__ void embw_kernel(const float* __restrict__ emb, const float* __restrict__ Wxh,
                            const float* __restrict__ bh, float* __restrict__ embWbh) {
  int j = threadIdx.x;  // 128 threads
  if (j >= 128) return;
  for (int o = 0; o < 3; ++o) {
    float s = bh[j];
    for (int d = 0; d < 128; ++d) s += emb[o * 128 + d] * Wxh[d * 128 + j];
    embWbh[o * 128 + j] = s;
  }
}

// ---------------------------------------------------------------------------
// Pack a row-major f32 (K x N) weight into the WMMA B-operand per-lane layout:
// out[((kc*NT + nt)*32 + lane)*16 + e] = (f16) W[(kc*32 + (e<8 ? (lane>>4)*8+e : 16+(lane>>4)*8+e-8))*N + nt*16 + (lane&15)]
// GEMM kernels then fetch one v16h (32B contiguous) per WMMA operand.
// ---------------------------------------------------------------------------
__global__ void packB_kernel(const float* __restrict__ W, _Float16* __restrict__ out,
                             int K, int N) {
  int i = blockIdx.x * blockDim.x + threadIdx.x;
  int NT = N >> 4;
  int total = (K >> 5) * NT * 32;
  if (i >= total) return;
  int lane = i & 31, t = i >> 5;
  int nt = t % NT, kc = t / NT;
  int h = lane >> 4, ln = lane & 15;
  int n = nt * 16 + ln;
  v16h bb;
#pragma unroll
  for (int e = 0; e < 16; ++e)
    bb[e] = (_Float16)W[(size_t)(kc * 32 + ((e < 8) ? (h * 8 + e) : (16 + h * 8 + (e - 8)))) * N + n];
  *(v16h*)(out + (size_t)i * 16) = bb;
}

// Same, for OIHW conv weights viewed as B[k=c*9+ky*3+kx][n=o]: K=288, N=32.
__global__ void packB_conv_kernel(const float* __restrict__ w, _Float16* __restrict__ out) {
  int i = blockIdx.x * blockDim.x + threadIdx.x;
  if (i >= 9 * 2 * 32) return;
  int lane = i & 31, t = i >> 5;
  int nt = t & 1, kc = t >> 1;
  int h = lane >> 4, ln = lane & 15;
  int n = nt * 16 + ln;
  v16h bb;
#pragma unroll
  for (int e = 0; e < 16; ++e) {
    int k = kc * 32 + ((e < 8) ? (h * 8 + e) : (16 + h * 8 + (e - 8)));
    int c = k / 9, r = k - c * 9;
    int ky = r / 3, kx = r - ky * 3;
    bb[e] = (_Float16)w[((n * 32 + c) * 3 + ky) * 3 + kx];
  }
  *(v16h*)(out + (size_t)i * 16) = bb;
}

// ---------------------------------------------------------------------------
// conv1 (Cin=1, 3x3 VALID) + BN + ReLU + AvgPool2, fused scalar kernel.
// x:(512,1,32,32) -> out:(512,32,15,15)
// ---------------------------------------------------------------------------
__global__ __launch_bounds__(256)
void conv1_kernel(const float* __restrict__ x, const float* __restrict__ w,
                  const float* __restrict__ ss, float* __restrict__ out) {
  int bc = blockIdx.x;           // 512*32
  int b = bc >> 5, c = bc & 31;
  int t = threadIdx.x;
  if (t >= 225) return;
  int py = t / 15, px = t % 15;
  const float* xin = x + b * 1024;
  float scale = ss[c], shift = ss[32 + c];
  float wc[9];
#pragma unroll
  for (int i = 0; i < 9; ++i) wc[i] = w[c * 9 + i];
  float acc = 0.0f;
#pragma unroll
  for (int dy = 0; dy < 2; ++dy)
#pragma unroll
    for (int dx = 0; dx < 2; ++dx) {
      int oy = 2 * py + dy, ox = 2 * px + dx;
      float s = 0.0f;
#pragma unroll
      for (int ky = 0; ky < 3; ++ky)
#pragma unroll
        for (int kx = 0; kx < 3; ++kx)
          s += xin[(oy + ky) * 32 + ox + kx] * wc[ky * 3 + kx];
      acc += fmaxf(s * scale + shift, 0.0f);
    }
  out[bc * 225 + t] = acc * 0.25f;
}

// ---------------------------------------------------------------------------
// Implicit-GEMM conv (Cin=32, 3x3 VALID) + BN + ReLU via WMMA f16.
// M = B*(H-2)*(W-2), N = 32, K = 288. H/W/M compile-time -> all div/mod are
// constant magic-multiplies. B operand pre-packed f16 (one b128x2 per WMMA).
// ---------------------------------------------------------------------------
template <int H, int W, int M>
__global__ __launch_bounds__(256)
void conv_wmma_kernel(const float* __restrict__ in, const _Float16* __restrict__ Bp,
                      const float* __restrict__ ss, float* __restrict__ out) {
  constexpr int Cin = 32, OH = H - 2, OW = W - 2, plane = OH * OW;
  int wave = threadIdx.x >> 5, lane = threadIdx.x & 31;
  int m0 = (blockIdx.x * 8 + wave) * 16;
  if (m0 >= M) return;
  int h = lane >> 4, ln = lane & 15;

  int mA = m0 + ln;
  bool mok = (mA < M);
  int mAc = mok ? mA : 0;
  int bA = mAc / plane, remA = mAc % plane;
  int oyA = remA / OW, oxA = remA % OW;
  const float* inA = in + ((size_t)bA * Cin) * H * W + oyA * W + oxA;
  __builtin_prefetch(inA, 0, 1);

  v8f z = {};
  v8f acc0 = z, acc1 = z;
#pragma unroll
  for (int kc = 0; kc < 9; ++kc) {
    int k0 = kc * 32;
    v16h a;
#pragma unroll
    for (int e = 0; e < 16; ++e) {
      int k = k0 + ((e < 8) ? (h * 8 + e) : (16 + h * 8 + (e - 8)));
      int c = k / 9, t = k - c * 9;
      int ky = t / 3, kx = t - ky * 3;
      float v = mok ? inA[(c * H + ky) * W + kx] : 0.0f;
      a[e] = (_Float16)v;
    }
    v16h b0 = *(const v16h*)(Bp + (size_t)((kc * 2 + 0) * 32 + lane) * 16);
    v16h b1 = *(const v16h*)(Bp + (size_t)((kc * 2 + 1) * 32 + lane) * 16);
    acc0 = WMMA_F32_F16(a, b0, acc0);
    acc1 = WMMA_F32_F16(a, b1, acc1);
  }
#pragma unroll
  for (int e = 0; e < 8; ++e) {
    int m = m0 + h * 8 + e;
    if (m >= M) continue;
    int b = m / plane, rem = m % plane;
    int oy = rem / OW, ox = rem % OW;
#pragma unroll
    for (int nt = 0; nt < 2; ++nt) {
      int n = nt * 16 + ln;
      float v = (nt ? acc1[e] : acc0[e]) * ss[n] + ss[32 + n];
      out[((b * 32 + n) * OH + oy) * OW + ox] = fmaxf(v, 0.0f);
    }
  }
}

// AvgPool2d(2) on (BC, Hi, Wi) -> (BC, Ho, Wo)
__global__ void pool_kernel(const float* __restrict__ in, float* __restrict__ out,
                            int Hi, int Wi, int Ho, int Wo, int total) {
  int i = blockIdx.x * blockDim.x + threadIdx.x;
  if (i >= total) return;
  int px = i % Wo, t = i / Wo;
  int py = t % Ho, bc = t / Ho;
  const float* p = in + ((size_t)bc * Hi + 2 * py) * Wi + 2 * px;
  out[i] = 0.25f * (p[0] + p[1] + p[Wi] + p[Wi + 1]);
}

// hr/hs init: broadcast feat (512,128) to (512*8,128)
__global__ void bcast_kernel(const float* __restrict__ feat, float* __restrict__ hr,
                             float* __restrict__ hs) {
  int i = blockIdx.x * blockDim.x + threadIdx.x;
  if (i >= 4096 * 128) return;
  int d = i & 127, r = i >> 7;
  float v = feat[(r >> 3) * 128 + d];
  hr[i] = v;
  hs[i] = v;
}

// Pack one 16-row x 32-col f32 A tile slice into the WMMA A-operand v16h:
// lane ln row, halves give K = h*8+e (e<8) and 16+h*8+(e-8); both runs contiguous.
__device__ __forceinline__ v16h pack_a_rowmajor(const float* __restrict__ rowbase,
                                                int k0, int h) {
  const float* p = rowbase + k0 + h * 8;
  f32x4 r0 = *(const f32x4*)(p);
  f32x4 r1 = *(const f32x4*)(p + 4);
  f32x4 r2 = *(const f32x4*)(p + 16);
  f32x4 r3 = *(const f32x4*)(p + 20);
  v16h a;
#pragma unroll
  for (int e = 0; e < 4; ++e) {
    a[e]      = (_Float16)r0[e];
    a[4 + e]  = (_Float16)r1[e];
    a[8 + e]  = (_Float16)r2[e];
    a[12 + e] = (_Float16)r3[e];
  }
  return a;
}

// ---------------------------------------------------------------------------
// One beam-search step: logits -> log_softmax(T=0.4) -> top-8 of 24 ->
// beam gather of hr and op history.  One wave per sample.
// ---------------------------------------------------------------------------
__global__ __launch_bounds__(32)
void router_kernel(const float* __restrict__ hr, const float* __restrict__ Who,
                   const float* __restrict__ bo, const float* __restrict__ sc_in,
                   const int* __restrict__ ops_in, float* __restrict__ sc_out,
                   int* __restrict__ ops_out, float* __restrict__ hrg, int depth) {
  int b = blockIdx.x, lane = threadIdx.x;
  __shared__ float lt[24];
  __shared__ float tot[24];
  __shared__ int sel[8];
  __shared__ int bsel[8];
  if (lane < 24) {
    int beam = lane / 3, o = lane - beam * 3;
    const float* hv = hr + ((size_t)b * 8 + beam) * 128;
    float s = bo[o];
    for (int d = 0; d < 128; ++d) s += hv[d] * Who[d * 3 + o];
    lt[lane] = s * 2.5f;  // /TEMP, TEMP=0.4
  }
  __syncthreads();
  if (lane < 24) {
    int beam = lane / 3;
    float m = fmaxf(lt[beam * 3], fmaxf(lt[beam * 3 + 1], lt[beam * 3 + 2]));
    float se = expf(lt[beam * 3] - m) + expf(lt[beam * 3 + 1] - m) + expf(lt[beam * 3 + 2] - m);
    float logp = lt[lane] - m - logf(se);
    float sc = (depth == 0) ? ((beam == 0) ? 0.0f : -1e9f) : sc_in[b * 8 + beam];
    tot[lane] = sc + logp;
  }
  __syncthreads();
  if (lane == 0) {  // top-8, ties -> lowest index (lax.top_k semantics)
    unsigned used = 0;
    for (int i = 0; i < 8; ++i) {
      int best = 0;
      float bv = -3.4e38f;
      for (int j = 0; j < 24; ++j) {
        if (used & (1u << j)) continue;
        if (tot[j] > bv) { bv = tot[j]; best = j; }
      }
      used |= 1u << best;
      sel[i] = best;
      sc_out[b * 8 + i] = bv;
    }
  }
  __syncthreads();
  if (lane < 8) {
    int idx = sel[lane];
    int bi = idx / 3, o = idx - bi * 3;
    bsel[lane] = bi;
    for (int dd = 0; dd < depth; ++dd)
      ops_out[dd * 4096 + b * 8 + lane] = ops_in[dd * 4096 + b * 8 + bi];
    ops_out[depth * 4096 + b * 8 + lane] = o;
  }
  __syncthreads();
  for (int j = lane; j < 8 * 128; j += 32) {
    int k = j >> 7, d = j & 127;
    hrg[((size_t)b * 8 + k) * 128 + d] = hr[((size_t)b * 8 + bsel[k]) * 128 + d];
  }
}

// ---------------------------------------------------------------------------
// hr_out = tanh(hr_g @ Whh + embWbh[op]) : M=4096, K=128, N=128 via WMMA.
// B pre-packed f16; A loaded as 4x b128 + cvt.
// ---------------------------------------------------------------------------
__global__ __launch_bounds__(256)
void rnn_update_kernel(const float* __restrict__ hrg, const _Float16* __restrict__ WhhP,
                       const float* __restrict__ embWbh, const int* __restrict__ opd,
                       float* __restrict__ hro) {
  int wave = threadIdx.x >> 5, lane = threadIdx.x & 31;
  int m0 = (blockIdx.x * 8 + wave) * 16;
  int h = lane >> 4, ln = lane & 15;
  const float* rowbase = hrg + (size_t)(m0 + ln) * 128;
  __builtin_prefetch(rowbase, 0, 1);
  v8f z = {};
  v8f acc[8];
#pragma unroll
  for (int i = 0; i < 8; ++i) acc[i] = z;
#pragma unroll
  for (int kc = 0; kc < 4; ++kc) {
    v16h a = pack_a_rowmajor(rowbase, kc * 32, h);
#pragma unroll
    for (int nt = 0; nt < 8; ++nt) {
      v16h bb = *(const v16h*)(WhhP + (size_t)((kc * 8 + nt) * 32 + lane) * 16);
      acc[nt] = WMMA_F32_F16(a, bb, acc[nt]);
    }
  }
#pragma unroll
  for (int e = 0; e < 8; ++e) {
    int m = m0 + h * 8 + e;
    int o = opd[m];
#pragma unroll
    for (int nt = 0; nt < 8; ++nt) {
      int n = nt * 16 + ln;
      hro[(size_t)m * 128 + n] = tanhf(acc[nt][e] + embWbh[o * 128 + n]);
    }
  }
}

// ---------------------------------------------------------------------------
// Grouped expert GEMM: hs_out[r] = relu(hs_in[r] @ Wexp[o[r]] + bexp[o[r]]).
// 3 passes (one per expert) per tile, rows written selectively.
// ---------------------------------------------------------------------------
__global__ __launch_bounds__(256)
void expert_wmma_kernel(const float* __restrict__ hs_in, const _Float16* __restrict__ WexpP,
                        const float* __restrict__ bexp, const int* __restrict__ od,
                        float* __restrict__ hs_out) {
  int wave = threadIdx.x >> 5, lane = threadIdx.x & 31;
  int m0 = (blockIdx.x * 8 + wave) * 16;
  int h = lane >> 4, ln = lane & 15;
  const float* rowbase = hs_in + (size_t)(m0 + ln) * 128;
  __builtin_prefetch(rowbase, 0, 1);
  v8f z = {};
  for (int e0 = 0; e0 < 3; ++e0) {
    const _Float16* Bp = WexpP + (size_t)e0 * 128 * 128;
    v8f acc[8];
#pragma unroll
    for (int i = 0; i < 8; ++i) acc[i] = z;
#pragma unroll
    for (int kc = 0; kc < 4; ++kc) {
      v16h a = pack_a_rowmajor(rowbase, kc * 32, h);
#pragma unroll
      for (int nt = 0; nt < 8; ++nt) {
        v16h bb = *(const v16h*)(Bp + (size_t)((kc * 8 + nt) * 32 + lane) * 16);
        acc[nt] = WMMA_F32_F16(a, bb, acc[nt]);
      }
    }
#pragma unroll
    for (int e = 0; e < 8; ++e) {
      int m = m0 + h * 8 + e;
      if (od[m] != e0) continue;
#pragma unroll
      for (int nt = 0; nt < 8; ++nt) {
        int n = nt * 16 + ln;
        hs_out[(size_t)m * 128 + n] = fmaxf(acc[nt][e] + bexp[e0 * 128 + n], 0.0f);
      }
    }
  }
}

// out = log_softmax(hs @ Wout + bout) : (4096,10), one thread per row.
__global__ void head_kernel(const float* __restrict__ hs, const float* __restrict__ Wout,
                            const float* __restrict__ bout, float* __restrict__ out) {
  int r = blockIdx.x * blockDim.x + threadIdx.x;
  if (r >= 4096) return;
  float lg[10];
#pragma unroll
  for (int j = 0; j < 10; ++j) lg[j] = bout[j];
  for (int d = 0; d < 128; ++d) {
    float hv = hs[(size_t)r * 128 + d];
#pragma unroll
    for (int j = 0; j < 10; ++j) lg[j] += hv * Wout[d * 10 + j];
  }
  float m = lg[0];
#pragma unroll
  for (int j = 1; j < 10; ++j) m = fmaxf(m, lg[j]);
  float s = 0.0f;
#pragma unroll
  for (int j = 0; j < 10; ++j) s += expf(lg[j] - m);
  float ls = logf(s);
#pragma unroll
  for (int j = 0; j < 10; ++j) out[(size_t)r * 10 + j] = lg[j] - m - ls;
}

// ---------------------------------------------------------------------------
extern "C" void kernel_launch(void* const* d_in, const int* in_sizes, int n_in,
                              void* d_out, int out_size, void* d_ws, size_t ws_size,
                              hipStream_t stream) {
  (void)in_sizes; (void)n_in; (void)out_size; (void)ws_size;
  const float* x       = (const float*)d_in[0];
  const float* conv1_w = (const float*)d_in[1];
  const float* conv1_b = (const float*)d_in[2];
  const float* bn1_g   = (const float*)d_in[3];
  const float* bn1_b   = (const float*)d_in[4];
  const float* bn1_m   = (const float*)d_in[5];
  const float* bn1_v   = (const float*)d_in[6];
  const float* conv2_w = (const float*)d_in[7];
  const float* conv2_b = (const float*)d_in[8];
  const float* bn2_g   = (const float*)d_in[9];
  const float* bn2_b   = (const float*)d_in[10];
  const float* bn2_m   = (const float*)d_in[11];
  const float* bn2_v   = (const float*)d_in[12];
  const float* conv3_w = (const float*)d_in[13];
  const float* conv3_b = (const float*)d_in[14];
  const float* bn3_g   = (const float*)d_in[15];
  const float* bn3_b   = (const float*)d_in[16];
  const float* bn3_m   = (const float*)d_in[17];
  const float* bn3_v   = (const float*)d_in[18];
  const float* Wxh     = (const float*)d_in[19];
  const float* Whh     = (const float*)d_in[20];
  const float* bh      = (const float*)d_in[21];
  const float* Who     = (const float*)d_in[22];
  const float* bo      = (const float*)d_in[23];
  const float* emb     = (const float*)d_in[24];
  const float* Wexp    = (const float*)d_in[25];
  const float* bexp    = (const float*)d_in[26];
  const float* Wout    = (const float*)d_in[27];
  const float* bout    = (const float*)d_in[28];

  // workspace carve-out (all offsets kept 32B-aligned)
  float* ws = (float*)d_ws;
  size_t off = 0;
  auto alloc = [&](size_t n) { float* p = ws + off; off += (n + 7) & ~(size_t)7; return p; };
  float* out1   = alloc((size_t)512 * 32 * 15 * 15);  // conv1+pool
  float* c2     = alloc((size_t)512 * 32 * 13 * 13);  // conv2 pre-pool
  float* p2     = alloc((size_t)512 * 32 * 6 * 6);
  float* c3     = alloc((size_t)512 * 32 * 4 * 4);    // conv3 pre-pool
  float* feat   = alloc((size_t)512 * 128);           // pool3 == feat
  float* hr_a   = alloc((size_t)4096 * 128);
  float* hr_b   = alloc((size_t)4096 * 128);
  float* hs_a   = alloc((size_t)4096 * 128);
  float* hs_b   = alloc((size_t)4096 * 128);
  float* embWbh = alloc(3 * 128);
  float* ss1 = alloc(64);
  float* ss2 = alloc(64);
  float* ss3 = alloc(64);
  float* sc_a = alloc(4096);
  float* sc_b = alloc(4096);
  int* ops_a = (int*)alloc(3 * 4096);
  int* ops_b = (int*)alloc(3 * 4096);
  _Float16* WhhP  = (_Float16*)alloc(16384 / 2);       // 4*8*32*16 f16
  _Float16* WexpP = (_Float16*)alloc(3 * 16384 / 2);   // 3 experts
  _Float16* c2wP  = (_Float16*)alloc(9216 / 2);        // 9*2*32*16 f16
  _Float16* c3wP  = (_Float16*)alloc(9216 / 2);

  // small precomputes + weight packing (run once, reused by every tile)
  bnfold_kernel<<<1, 32, 0, stream>>>(bn1_g, bn1_b, bn1_m, bn1_v, conv1_b, ss1);
  bnfold_kernel<<<1, 32, 0, stream>>>(bn2_g, bn2_b, bn2_m, bn2_v, conv2_b, ss2);
  bnfold_kernel<<<1, 32, 0, stream>>>(bn3_g, bn3_b, bn3_m, bn3_v, conv3_b, ss3);
  embw_kernel<<<1, 128, 0, stream>>>(emb, Wxh, bh, embWbh);
  packB_kernel<<<4, 256, 0, stream>>>(Whh, WhhP, 128, 128);
  for (int e = 0; e < 3; ++e)
    packB_kernel<<<4, 256, 0, stream>>>(Wexp + (size_t)e * 16384, WexpP + (size_t)e * 16384,
                                        128, 128);
  packB_conv_kernel<<<3, 256, 0, stream>>>(conv2_w, c2wP);
  packB_conv_kernel<<<3, 256, 0, stream>>>(conv3_w, c3wP);

  // backbone
  conv1_kernel<<<512 * 32, 256, 0, stream>>>(x, conv1_w, ss1, out1);
  // conv2: M = 512*13*13 = 86528 -> 5408 tiles -> 676 blocks of 8 waves
  conv_wmma_kernel<15, 15, 86528><<<676, 256, 0, stream>>>(out1, c2wP, ss2, c2);
  pool_kernel<<<(589824 + 255) / 256, 256, 0, stream>>>(c2, p2, 13, 13, 6, 6, 589824);
  // conv3: M = 512*4*4 = 8192 -> 512 tiles -> 64 blocks
  conv_wmma_kernel<6, 6, 8192><<<64, 256, 0, stream>>>(p2, c3wP, ss3, c3);
  pool_kernel<<<(65536 + 255) / 256, 256, 0, stream>>>(c3, feat, 4, 4, 2, 2, 65536);

  // beam-search router (DEPTH=3)
  bcast_kernel<<<2048, 256, 0, stream>>>(feat, hr_a, hs_a);
  float* sc_in = sc_b; float* sc_out = sc_a;
  int* ops_in = ops_b; int* ops_out = ops_a;
  for (int d = 0; d < 3; ++d) {
    router_kernel<<<512, 32, 0, stream>>>(hr_a, Who, bo, sc_in, ops_in, sc_out, ops_out, hr_b, d);
    rnn_update_kernel<<<32, 256, 0, stream>>>(hr_b, WhhP, embWbh, ops_out + d * 4096, hr_a);
    float* tf = sc_in; sc_in = sc_out; sc_out = tf;
    int* ti = ops_in; ops_in = ops_out; ops_out = ti;
  }
  const int* ops_final = ops_in;  // after 3 swaps, final gathered history

  // routed expert chain
  float* src = hs_a; float* dst = hs_b;
  for (int step = 0; step < 3; ++step) {
    expert_wmma_kernel<<<32, 256, 0, stream>>>(src, WexpP, bexp, ops_final + step * 4096, dst);
    float* t = src; src = dst; dst = t;
  }

  // head
  head_kernel<<<16, 256, 0, stream>>>(src, Wout, bout, (float*)d_out);
}
